// CausalTransformerLM_3539053052679
// MI455X (gfx1250) — compile-verified, matrix operand order checked by reading, not run on hardware
//
#include <hip/hip_runtime.h>

// ---------------------------------------------------------------------------
// CDNA5 (gfx1250) causal transformer LM forward.
//  - All matmuls: v_wmma_f32_16x16x32_f16 (f16 in, f32 accumulate).
//  - GEMM tiles staged with CDNA5 async data movers, double-buffered:
//      A tile: GLOBAL_LOAD_ASYNC_TO_LDS_B128 (ASYNCcnt, inline asm)
//      B tile: TENSOR_LOAD_TO_LDS (TDM, TENSORcnt) with LDS pad feature
//              reproducing the 40-half padded row layout (fallback: async).
//  - Flash-style attention (online softmax), no LxL score buffer.
// ---------------------------------------------------------------------------

#define B_   2
#define L_   1024
#define D_   768
#define H_   12
#define HD_  64
#define FF_  3072
#define V_   32000
#define NL_  2
#define BOS_ 1
#define SEP_ 2
#define WIN_ 512

typedef __attribute__((ext_vector_type(16))) _Float16 v16h;
typedef __attribute__((ext_vector_type(8)))  float    v8f;

#if defined(__has_builtin)
# if __has_builtin(__builtin_amdgcn_tensor_load_to_lds)
#  define USE_TDM 1
# else
#  define USE_TDM 0
# endif
#else
# define USE_TDM 0
#endif

// Low 32 bits of a generic pointer into LDS == wave-relative LDS byte address
// (flat LDS aperture, ISA 10.2).
__device__ __forceinline__ unsigned lds_off32(const void* p) {
  return (unsigned)(unsigned long long)p;
}

// GLOBAL_LOAD_ASYNC_TO_LDS_B128, GVS mode: LDS byte address taken from vdst;
// mem = SADDR(64) + VADDR(32, unsigned byte offset). Tracked by ASYNCcnt.
__device__ __forceinline__ void async_load_b128(unsigned lds_addr,
                                                const void* sbase,
                                                unsigned byte_off) {
  asm volatile("global_load_async_to_lds_b128 %0, %1, %2"
               :: "v"(lds_addr), "v"(byte_off), "s"(sbase)
               : "memory");
}

template <int N>
__device__ __forceinline__ void wait_asynccnt() {
  asm volatile("s_wait_asynccnt %0" :: "n"(N) : "memory");
}

#if USE_TDM
typedef __attribute__((ext_vector_type(4))) unsigned int tdm_u4;
typedef __attribute__((ext_vector_type(8))) int          tdm_i8;
typedef __attribute__((ext_vector_type(4))) int          tdm_i4;

// One 2D tile: 128 rows x 32 halves (64B), row stride = K halves, into LDS
// with pad_interval=16 DWORDs / pad_amount=4 DWORDs => 40-half padded rows.
__device__ __forceinline__ void tdm_load_wtile(const _Float16* gsrc,
                                               unsigned lds_addr, int K) {
  const unsigned long long ga = (unsigned long long)gsrc;
  tdm_u4 g0;
  g0[0] = 1u;                                  // count=1, user-mode descriptor
  g0[1] = lds_addr;                            // lds_addr [63:32]
  g0[2] = (unsigned)ga;                        // global_addr [95:64]
  g0[3] = (unsigned)((ga >> 32) & 0x01ffffffu) // global_addr [120:96]
          | 0x80000000u;                       // type=2 ("image") [127:126]
  tdm_i8 g1;
  g1[0] = (int)((1u << 16)                     // data_size = 2 bytes
                | (1u << 20)                   // pad_enable
                | (3u << 22)                   // pad_interval: 16 DWORDs
                | (3u << 25));                 // pad_amount: 4 DWORDs
  g1[1] = (int)(((unsigned)K & 0xffffu) << 16);      // tensor_dim0[15:0]
  g1[2] = (int)(((unsigned)K >> 16) | (128u << 16)); // dim0[31:16] | dim1[15:0]
  g1[3] = (int)(32u << 16);                    // dim1[31:16]=0 | tile_dim0=32
  g1[4] = 128;                                 // tile_dim1=128 | tile_dim2=0
  g1[5] = K;                                   // tensor_dim0_stride[31:0]
  g1[6] = 0;
  g1[7] = 0;
  tdm_i4 z4 = {0, 0, 0, 0};
  tdm_i8 z8 = {0, 0, 0, 0, 0, 0, 0, 0};
  // 6-arg form (this toolchain): (g0, g1, g2, g3, g4, cpol)
  __builtin_amdgcn_tensor_load_to_lds(g0, g1, z4, z4, z8, 0);
}
#endif

// Fragment loader per ISA 7.12.2 (16-bit A, 16x32): lane holds row m=lane&15;
// lanes 0-15 carry K 0..7 / 16..23, lanes 16-31 carry K 8..15 / 24..31.
// B (32x16) mirrors it, so W row-major == B column-major (computes A @ W^T).
__device__ __forceinline__ v16h load_frag(const _Float16* rowbase, int kb) {
  union { v16h h; uint4 u[2]; } f;
  f.u[0] = *reinterpret_cast<const uint4*>(rowbase + kb);
  f.u[1] = *reinterpret_cast<const uint4*>(rowbase + 16 + kb);
  return f.h;
}

__device__ __forceinline__ v8f wmma_f16(v16h a, v16h b, v8f c) {
  return __builtin_amdgcn_wmma_f32_16x16x32_f16(false, a, false, b, (short)0, c,
                                                false, false);
}

// ------------------------------- GEMM --------------------------------------
#define GF_BIAS 1
#define GF_GELU 2
#define GF_RES  4
#define GF_F32  8
#define GF_F16  16

template <int FLAGS>
__global__ void __launch_bounds__(256)
gemm_wmma_kernel(const _Float16* __restrict__ A, const _Float16* __restrict__ W,
                 int M, int N, int K,
                 const float* __restrict__ bias, const float* __restrict__ resid,
                 float* __restrict__ outF, _Float16* __restrict__ outH) {
  constexpr int TM = 64, TN = 128, TK = 32, LDSS = TK + 8;  // 40-half rows
  __shared__ __align__(16) _Float16 As[2][TM][LDSS];
  __shared__ __align__(16) _Float16 Bs[2][TN][LDSS];
  const int t = threadIdx.x, lane = t & 31, wid = t >> 5;
  const int bm = blockIdx.y, bn = blockIdx.x;
  const int wm = wid >> 2, wn = wid & 3;                    // 2x4 wave grid
  const int mrow = lane & 15, kb = (lane < 16) ? 0 : 8, hi = lane >> 4;
  const int arow = t >> 2, acol = (t & 3) * 8;              // A: 1 b128/thread

  auto issue_tile = [&](int k0, int buf) {
    async_load_b128(lds_off32(&As[buf][arow][acol]), A,
                    (unsigned)(((unsigned)(bm * TM + arow) * (unsigned)K + k0 + acol) * 2u));
#if USE_TDM
    if (wid == 0)  // one TDM descriptor moves the whole 128x32 weight tile
      tdm_load_wtile(W + (size_t)(bn * TN) * K + k0,
                     lds_off32(&Bs[buf][0][0]), K);
#else
#pragma unroll
    for (int i = 0; i < 2; ++i) {
      const int c = t * 2 + i, row = c >> 2, col = (c & 3) * 8;
      async_load_b128(lds_off32(&Bs[buf][row][col]), W,
                      (unsigned)(((unsigned)(bn * TN + row) * (unsigned)K + k0 + col) * 2u));
    }
#endif
  };

  v8f acc[2][2] = {};
  const int nk = K / TK;
  issue_tile(0, 0);
  for (int kt = 0; kt < nk; ++kt) {
    const int buf = kt & 1;
    if (kt + 1 < nk) {
      issue_tile((kt + 1) * TK, buf ^ 1);   // overlap fetch with WMMA
#if USE_TDM
      wait_asynccnt<1>();                   // in-order: tile kt landed
      if (wid == 0) __builtin_amdgcn_s_wait_tensorcnt(1);
#else
      wait_asynccnt<3>();
#endif
    } else {
      wait_asynccnt<0>();
#if USE_TDM
      if (wid == 0) __builtin_amdgcn_s_wait_tensorcnt(0);
#endif
    }
    __syncthreads();
    v16h af[2], bf[2];
#pragma unroll
    for (int ms = 0; ms < 2; ++ms)
      af[ms] = load_frag(&As[buf][wm * 32 + ms * 16 + mrow][0], kb);
#pragma unroll
    for (int ns = 0; ns < 2; ++ns)
      bf[ns] = load_frag(&Bs[buf][wn * 32 + ns * 16 + mrow][0], kb);
#pragma unroll
    for (int ms = 0; ms < 2; ++ms)
#pragma unroll
      for (int ns = 0; ns < 2; ++ns)
        acc[ms][ns] = wmma_f16(af[ms], bf[ns], acc[ms][ns]);
    __syncthreads();  // buffer kt&1 free for the kt+2 issue
  }
  // Epilogue. C layout: lane = n (&15), VGPR r = m row r + 8*(lane>>4).
#pragma unroll
  for (int ms = 0; ms < 2; ++ms)
#pragma unroll
    for (int ns = 0; ns < 2; ++ns) {
      const int n = bn * TN + wn * 32 + ns * 16 + mrow;
#pragma unroll
      for (int r = 0; r < 8; ++r) {
        const int m = bm * TM + wm * 32 + ms * 16 + r + 8 * hi;
        float v = acc[ms][ns][r];
        if (FLAGS & GF_BIAS) v += bias[n];
        if (FLAGS & GF_GELU) v = 0.5f * v * (1.0f + erff(v * 0.70710678118654752f));
        if (FLAGS & GF_RES)  v += resid[(size_t)m * N + n];
        if (FLAGS & GF_F32)  outF[(size_t)m * N + n] = v;
        if (FLAGS & GF_F16)  outH[(size_t)m * N + n] = (_Float16)v;
      }
    }
}

// ----------------------------- Attention -----------------------------------
// One wave per (b, h, 16-query tile). Keys in blocks of 32: S(16x32) via 4
// WMMAs, online softmax (16-lane shfl_xor reductions), P restaged through LDS
// into an A-fragment, O(16x64) += P @ V via 4 WMMAs against V^T (hd-major).
__global__ void __launch_bounds__(256)
attn_kernel(const _Float16* __restrict__ q16, const _Float16* __restrict__ k16,
            const _Float16* __restrict__ vT16, const int* __restrict__ seg,
            const int* __restrict__ gk, const int* __restrict__ valid,
            _Float16* __restrict__ y16) {
  constexpr int PST = 40;
  __shared__ __align__(16) _Float16 P[8][16 * PST];
  const int lane = threadIdx.x & 31, wid = threadIdx.x >> 5;
  const int task = blockIdx.x * 8 + wid;  // B*H*(L/16) tasks, exact fit
  const int nQT = L_ / 16;
  const int qt = task % nQT;
  const int bh = task / nQT;
  const int h = bh % H_, b = bh / H_;
  const _Float16* Qp = q16 + ((size_t)(b * H_ + h) * L_ + qt * 16) * HD_;
  const _Float16* Kp = k16 + (size_t)(b * H_ + h) * L_ * HD_;
  const _Float16* Vp = vT16 + (size_t)(b * H_ + h) * HD_ * L_;
  const int mrow = lane & 15, kb = (lane < 16) ? 0 : 8, hi = lane >> 4;
  v16h qf[2];
#pragma unroll
  for (int j = 0; j < 2; ++j) qf[j] = load_frag(Qp + mrow * HD_ + j * 32, kb);
  float mrun[8], lrun[8];
  v8f o[4] = {};
  int iglob[8], segI[8];
#pragma unroll
  for (int r = 0; r < 8; ++r) {
    mrun[r] = -1e30f; lrun[r] = 0.0f;
    iglob[r] = qt * 16 + r + 8 * hi;
    segI[r] = seg[b * L_ + iglob[r]];
  }
  _Float16* Pw = &P[wid][0];
  const float SCALE = 0.125f;  // 1/sqrt(HD)
  const int nKB = qt / 2 + 1;
  for (int blk = 0; blk < nKB; ++blk) {
    const int j0 = blk * 32;
    v8f s0 = {}, s1 = {};
    const int kr0 = j0 + mrow, kr1 = j0 + 16 + mrow;
#pragma unroll
    for (int j = 0; j < 2; ++j) {
      v16h b0 = load_frag(Kp + (size_t)kr0 * HD_ + j * 32, kb);
      v16h b1 = load_frag(Kp + (size_t)kr1 * HD_ + j * 32, kb);
      s0 = wmma_f16(qf[j], b0, s0);
      s1 = wmma_f16(qf[j], b1, s1);
    }
    const int jg0 = j0 + mrow, jg1 = jg0 + 16;
    const int sj0 = seg[b * L_ + jg0], sj1 = seg[b * L_ + jg1];
    const int gj0 = gk[b * L_ + jg0], gj1 = gk[b * L_ + jg1];
    const int vj0 = valid[b * L_ + jg0], vj1 = valid[b * L_ + jg1];
#pragma unroll
    for (int r = 0; r < 8; ++r) {
      const int i = iglob[r];
      const bool a0 = vj0 && (jg0 <= i) && ((sj0 == segI[r]) || gj0 || ((i - jg0) <= WIN_));
      const bool a1 = vj1 && (jg1 <= i) && ((sj1 == segI[r]) || gj1 || ((i - jg1) <= WIN_));
      float x0 = a0 ? s0[r] * SCALE : -1e30f;  // finite sentinel: no NaN paths
      float x1 = a1 ? s1[r] * SCALE : -1e30f;
      float tm = fmaxf(x0, x1);
      tm = fmaxf(tm, __shfl_xor(tm, 1, 32));
      tm = fmaxf(tm, __shfl_xor(tm, 2, 32));
      tm = fmaxf(tm, __shfl_xor(tm, 4, 32));
      tm = fmaxf(tm, __shfl_xor(tm, 8, 32));
      const float mn = fmaxf(mrun[r], tm);
      const float p0 = __expf(x0 - mn), p1 = __expf(x1 - mn);
      const float corr = __expf(mrun[r] - mn);
      float rsum = p0 + p1;
      rsum += __shfl_xor(rsum, 1, 32);
      rsum += __shfl_xor(rsum, 2, 32);
      rsum += __shfl_xor(rsum, 4, 32);
      rsum += __shfl_xor(rsum, 8, 32);
      lrun[r] = lrun[r] * corr + rsum;
      mrun[r] = mn;
#pragma unroll
      for (int c = 0; c < 4; ++c) o[c][r] *= corr;
      Pw[(r + 8 * hi) * PST + mrow] = (_Float16)p0;       // C-layout -> LDS
      Pw[(r + 8 * hi) * PST + 16 + mrow] = (_Float16)p1;  // (wave-ordered DS)
    }
    v16h pf = load_frag(Pw + mrow * PST, kb);  // re-read as A-fragment
#pragma unroll
    for (int c = 0; c < 4; ++c) {
      v16h vf = load_frag(Vp + (size_t)(c * 16 + mrow) * L_ + j0, kb);
      o[c] = wmma_f16(pf, vf, o[c]);
    }
  }
#pragma unroll
  for (int c = 0; c < 4; ++c)
#pragma unroll
    for (int r = 0; r < 8; ++r) {
      const int i = iglob[r];
      y16[((size_t)b * L_ + i) * D_ + h * HD_ + c * 16 + mrow] =
          (_Float16)(o[c][r] / lrun[r]);
    }
}

// ----------------------------- Helpers -------------------------------------
__global__ void __launch_bounds__(256)
ln_kernel(const float* __restrict__ x, const float* __restrict__ s,
          const float* __restrict__ bvec, _Float16* __restrict__ out) {
  const int row = blockIdx.x, t = threadIdx.x;
  const float* xr = x + (size_t)row * D_;
  const float v0 = xr[t], v1 = xr[t + 256], v2 = xr[t + 512];
  float sum = v0 + v1 + v2, sq = v0 * v0 + v1 * v1 + v2 * v2;
#pragma unroll
  for (int off = 16; off >= 1; off >>= 1) {
    sum += __shfl_down(sum, off, 32);
    sq += __shfl_down(sq, off, 32);
  }
  __shared__ float rs[8], rq[8];
  __shared__ float mu_s, rstd_s;
  if ((t & 31) == 0) { rs[t >> 5] = sum; rq[t >> 5] = sq; }
  __syncthreads();
  if (t == 0) {
    float S = 0.f, Q = 0.f;
    for (int i = 0; i < 8; ++i) { S += rs[i]; Q += rq[i]; }
    const float mu = S / (float)D_;
    mu_s = mu;
    rstd_s = rsqrtf(Q / (float)D_ - mu * mu + 1e-5f);
  }
  __syncthreads();
  const float mu = mu_s, rstd = rstd_s;
  out[(size_t)row * D_ + t] = (_Float16)((v0 - mu) * rstd * s[t] + bvec[t]);
  out[(size_t)row * D_ + t + 256] = (_Float16)((v1 - mu) * rstd * s[t + 256] + bvec[t + 256]);
  out[(size_t)row * D_ + t + 512] = (_Float16)((v2 - mu) * rstd * s[t + 512] + bvec[t + 512]);
}

__global__ void embed_kernel(const int* __restrict__ tokens, const int* __restrict__ types,
                             const float* __restrict__ tok_emb,
                             const float* __restrict__ type_emb,
                             const float* __restrict__ pos_emb, float* __restrict__ x) {
  const size_t idx = (size_t)blockIdx.x * 256 + threadIdx.x;  // < B*L*D
  const int d = (int)(idx % D_);
  const int bl = (int)(idx / D_);
  const int l = bl % L_;
  x[idx] = tok_emb[(size_t)tokens[bl] * D_ + d] +
           type_emb[(size_t)types[bl] * D_ + d] + pos_emb[(size_t)l * D_ + d];
}

__global__ void seg_kernel(const int* __restrict__ tokens, const int* __restrict__ mask,
                           int* __restrict__ seg, int* __restrict__ gk) {
  const int b = threadIdx.x;
  if (b >= B_) return;
  int acc = 0;
  for (int l = 0; l < L_; ++l) {
    const int tok = tokens[b * L_ + l];
    const int val = mask[b * L_ + l] != 0;
    const int is_sep = ((tok == SEP_) && val) ? 1 : 0;
    acc += is_sep;
    seg[b * L_ + l] = acc;
    gk[b * L_ + l] = (((tok == BOS_) && val) || is_sep) ? 1 : 0;
  }
}

__global__ void split_kernel(const _Float16* __restrict__ qkv, _Float16* __restrict__ q16,
                             _Float16* __restrict__ k16, _Float16* __restrict__ vT16) {
  const size_t idx = (size_t)blockIdx.x * 256 + threadIdx.x;  // < B*L*3D
  const int c = (int)(idx % (3 * D_));
  const int bl = (int)(idx / (3 * D_));
  const int l = bl % L_, b = bl / L_;
  const int which = c / D_;
  const int h = (c % D_) / HD_, hd = c % HD_;
  const _Float16 v = qkv[idx];
  if (which == 0)
    q16[((size_t)(b * H_ + h) * L_ + l) * HD_ + hd] = v;
  else if (which == 1)
    k16[((size_t)(b * H_ + h) * L_ + l) * HD_ + hd] = v;
  else
    vT16[((size_t)(b * H_ + h) * HD_ + hd) * L_ + l] = v;  // V^T for B-frags
}

__global__ void f32_to_f16_kernel(const float* __restrict__ src, _Float16* __restrict__ dst) {
  const size_t i = (size_t)blockIdx.x * 256 + threadIdx.x;
  dst[i] = (_Float16)src[i];
}

// ----------------------------- Launch --------------------------------------
extern "C" void kernel_launch(void* const* d_in, const int* in_sizes, int n_in,
                              void* d_out, int out_size, void* d_ws, size_t ws_size,
                              hipStream_t stream) {
  const int* tokens = (const int*)d_in[0];
  const int* types = (const int*)d_in[1];
  const int* attnmask = (const int*)d_in[2];
  const float* tok_emb = (const float*)d_in[3];
  const float* type_emb = (const float*)d_in[4];
  const float* pos_emb = (const float*)d_in[5];
  const float* qkv_w = (const float*)d_in[6];
  const float* out_w = (const float*)d_in[7];
  const float* ln1_s = (const float*)d_in[8];
  const float* ln1_b = (const float*)d_in[9];
  const float* ln2_s = (const float*)d_in[10];
  const float* ln2_b = (const float*)d_in[11];
  const float* ff_w1 = (const float*)d_in[12];
  const float* ff_b1 = (const float*)d_in[13];
  const float* ff_w2 = (const float*)d_in[14];
  const float* ff_b2 = (const float*)d_in[15];
  const float* lnf_s = (const float*)d_in[16];
  const float* lnf_b = (const float*)d_in[17];
  const float* head_w = (const float*)d_in[18];

  char* p = (char*)d_ws;
  auto carve = [&p](size_t bytes) -> void* {
    void* r = (void*)p;
    p += (bytes + 255) & ~(size_t)255;
    return r;
  };
  const size_t MTOK = (size_t)B_ * L_;  // 2048 rows
  float*    x      = (float*)carve(MTOK * D_ * 4);
  _Float16* h16    = (_Float16*)carve(MTOK * D_ * 2);
  _Float16* qkv16  = (_Float16*)carve(MTOK * 3 * D_ * 2);
  _Float16* q16    = (_Float16*)carve((size_t)B_ * H_ * L_ * HD_ * 2);
  _Float16* k16    = (_Float16*)carve((size_t)B_ * H_ * L_ * HD_ * 2);
  _Float16* vT16   = (_Float16*)carve((size_t)B_ * H_ * L_ * HD_ * 2);
  _Float16* y16    = (_Float16*)carve(MTOK * D_ * 2);
  _Float16* mid16  = (_Float16*)carve(MTOK * FF_ * 2);
  int*      seg    = (int*)carve(MTOK * 4);
  int*      gk     = (int*)carve(MTOK * 4);
  _Float16* qkvw16 = (_Float16*)carve((size_t)NL_ * 3 * D_ * D_ * 2);
  _Float16* outw16 = (_Float16*)carve((size_t)NL_ * D_ * D_ * 2);
  _Float16* ffw116 = (_Float16*)carve((size_t)NL_ * FF_ * D_ * 2);
  _Float16* ffw216 = (_Float16*)carve((size_t)NL_ * D_ * FF_ * 2);
  _Float16* headw16 = (_Float16*)carve((size_t)V_ * D_ * 2);

  f32_to_f16_kernel<<<(NL_ * 3 * D_ * D_) / 256, 256, 0, stream>>>(qkv_w, qkvw16);
  f32_to_f16_kernel<<<(NL_ * D_ * D_) / 256, 256, 0, stream>>>(out_w, outw16);
  f32_to_f16_kernel<<<(NL_ * FF_ * D_) / 256, 256, 0, stream>>>(ff_w1, ffw116);
  f32_to_f16_kernel<<<(NL_ * D_ * FF_) / 256, 256, 0, stream>>>(ff_w2, ffw216);
  f32_to_f16_kernel<<<(V_ * D_) / 256, 256, 0, stream>>>(head_w, headw16);

  seg_kernel<<<1, 32, 0, stream>>>(tokens, attnmask, seg, gk);
  embed_kernel<<<(int)((MTOK * D_) / 256), 256, 0, stream>>>(tokens, types, tok_emb,
                                                             type_emb, pos_emb, x);
  const int M = (int)MTOK;
  for (int l = 0; l < NL_; ++l) {
    ln_kernel<<<M, 256, 0, stream>>>(x, ln1_s + l * D_, ln1_b + l * D_, h16);
    gemm_wmma_kernel<GF_F16><<<dim3(3 * D_ / 128, M / 64), 256, 0, stream>>>(
        h16, qkvw16 + (size_t)l * 3 * D_ * D_, M, 3 * D_, D_, nullptr, nullptr,
        nullptr, qkv16);
    split_kernel<<<(int)((MTOK * 3 * D_) / 256), 256, 0, stream>>>(qkv16, q16, k16, vT16);
    attn_kernel<<<(B_ * H_ * (L_ / 16)) / 8, 256, 0, stream>>>(q16, k16, vT16, seg, gk,
                                                               attnmask, y16);
    gemm_wmma_kernel<GF_RES | GF_F32><<<dim3(D_ / 128, M / 64), 256, 0, stream>>>(
        y16, outw16 + (size_t)l * D_ * D_, M, D_, D_, nullptr, x, x, nullptr);
    ln_kernel<<<M, 256, 0, stream>>>(x, ln2_s + l * D_, ln2_b + l * D_, h16);
    gemm_wmma_kernel<GF_BIAS | GF_GELU | GF_F16>
        <<<dim3(FF_ / 128, M / 64), 256, 0, stream>>>(
            h16, ffw116 + (size_t)l * FF_ * D_, M, FF_, D_, ff_b1 + l * FF_, nullptr,
            nullptr, mid16);
    gemm_wmma_kernel<GF_BIAS | GF_RES | GF_F32>
        <<<dim3(D_ / 128, M / 64), 256, 0, stream>>>(
            mid16, ffw216 + (size_t)l * D_ * FF_, M, D_, FF_, ff_b2 + l * D_, x, x,
            nullptr);
  }
  ln_kernel<<<M, 256, 0, stream>>>(x, lnf_s, lnf_b, h16);
  gemm_wmma_kernel<GF_F32><<<dim3(V_ / 128, M / 64), 256, 0, stream>>>(
      h16, headw16, M, V_, D_, nullptr, nullptr, (float*)d_out, nullptr);
}
